// DIVeR_45629732553179
// MI455X (gfx1250) — matrix operand order repaired and checked by reading, not compiled, not run on hardware
//
#include <hip/hip_runtime.h>
#include <hip/hip_bf16.h>
#include <stdint.h>

// ---------------- problem constants ----------------
#define VOXEL_NUM   128
#define VOXEL_DIM   64
#define GRID_N1     129                   // VOXEL_NUM + 1
#define VOXEL_SIZE  0.015625f             // 2.0 / 128
#define INV_VS      64.0f                 // 1 / VOXEL_SIZE
#define XYZMIN_C    (-1.0f)
#define XYZMAX_C    (1.0f)
#define NRAYS       2048
#define MSEG        386                   // 3*(129)-1
#define NPLANES     387                   // 3*129
#define ROWS        790528                // NRAYS * MSEG
// d_out layout: color[ROWS*3] | sigma[ROWS] | mask[ROWS] | t_in[ROWS]
#define OFF_SIGMA   (ROWS * 3)
#define OFF_MASK    (ROWS * 4)
#define OFF_TIN     (ROWS * 5)

typedef _Float16 half_t;
typedef __attribute__((ext_vector_type(16))) _Float16 v16h;
typedef __attribute__((ext_vector_type(8)))  _Float16 v8h;
typedef __attribute__((ext_vector_type(8)))  float    v8f;

// ---------------- WMMA helpers (CDNA5 16x16x32 f16 -> f32) ----------------
__device__ __forceinline__ v8f wmma_f16(v16h a, v16h b, v8f c) {
    // D = A(16x32 f16) * B(32x16 f16) + C(16x16 f32)
    return __builtin_amdgcn_wmma_f32_16x16x32_f16(
        /*neg_a=*/false, a, /*neg_b=*/false, b,
        /*c_mod=*/(short)0, c, /*reuse_a=*/false, /*reuse_b=*/false);
}

// A fragment from LDS activation tile [16 rows x 64 halves], row stride 64.
// Lane layout (ISA 16-bit A 16x32): lanes 0-15 -> M=lane, elems 0..7: K=kh*8+e,
// elems 8..15: K=16+kh*8+(e-8), where kh = lane>>4 shifts K by 8.
// Both 8-half chunks are contiguous & 16B-aligned -> 2x ds_load_b128.
__device__ __forceinline__ v16h load_a(const half_t* act, int k0, int lane) {
    int m  = lane & 15;
    int kh = lane >> 4;
    const half_t* p = act + m * 64 + k0 + kh * 8;
    v8h lo = *(const v8h*)p;
    v8h hi = *(const v8h*)(p + 16);
    v16h f;
#pragma unroll
    for (int e = 0; e < 8; ++e) { f[e] = lo[e]; f[8 + e] = hi[e]; }
    return f;
}

// B fragment from LDS in PRE-SHUFFLED fragment order:
// frag*512 + lane*16 + e, where elem e holds K = ktile*32 + (lane>>4)*8 + (e<8 ? e : e+8),
// N = ntile*16 + (lane&15). Contiguous 32B per lane -> 2x ds_load_b128.
__device__ __forceinline__ v16h load_bf(const half_t* W, int frag, int lane) {
    return *(const v16h*)(W + (frag << 9) + lane * 16);
}

// Stage a weight matrix (f32, row-major [validK x srcCols]) into LDS in
// B-fragment order with zero padding. Fragments: ktile-major over ntiles:
// frag = ktile*ntiles + ntile, ktile in {0,1} (K chunks 0..31 / 32..63).
__device__ __forceinline__ void stage_frags(half_t* dst, const float* src,
                                            int srcCols, int validK, int validN,
                                            int ntiles, int tid) {
    const int total = 2 * ntiles * 512;
    for (int i = tid; i < total; i += 256) {
        int frag  = i >> 9;
        int r     = i & 511;
        int lw    = r >> 4;      // writer "lane" 0..31
        int e     = r & 15;      // element 0..15
        int ktile = frag / ntiles;
        int ntile = frag - ktile * ntiles;
        int k = ktile * 32 + (lw >> 4) * 8 + (e < 8 ? e : e + 8);
        int n = ntile * 16 + (lw & 15);
        float v = (k < validK && n < validN) ? src[k * srcCols + n] : 0.0f;
        dst[i] = (half_t)v;
    }
}

// Dense 64->64 layer, in-place on a per-wave LDS tile:
// act(16x64) = relu?(act @ W + b). W is fragment-ordered with ntiles=4.
// C/D layout: elem r -> M = r + 8*(lane>>4), N = ntile*16 + (lane&15).
__device__ __forceinline__ void dense64(half_t* act, const half_t* W, const float* bias,
                                        bool relu, int lane) {
    v16h a0 = load_a(act, 0, lane);
    v16h a1 = load_a(act, 32, lane);
    int n16 = lane & 15;
    int kh  = lane >> 4;
#pragma unroll
    for (int j = 0; j < 4; ++j) {
        v8f c = {};
        c = wmma_f16(a0, load_bf(W, j,     lane), c);
        c = wmma_f16(a1, load_bf(W, 4 + j, lane), c);
        float bv = bias[16 * j + n16];
#pragma unroll
        for (int r = 0; r < 8; ++r) {
            float v = c[r] + bv;
            if (relu) v = fmaxf(v, 0.0f);
            act[(r + 8 * kh) * 64 + 16 * j + n16] = (half_t)v;
        }
    }
}

// =====================================================================
// Kernel 1: per-ray geometry: plane t's, bitonic sort, Simpson-weighted
// trilinear voxel features -> f16 workspace; mask & t_in -> d_out.
// One block (256 thr) per ray.
// =====================================================================
__global__ void __launch_bounds__(256) geom_kernel(
    const float* __restrict__ os, const float* __restrict__ ds,
    const float* __restrict__ voxels,
    half_t* __restrict__ feat16, float* __restrict__ maskOut,
    float* __restrict__ tinOut) {

    __shared__ float ts[512];
    const int ray = blockIdx.x;
    const int tid = threadIdx.x;

    const float o0 = os[ray * 3 + 0], o1 = os[ray * 3 + 1], o2 = os[ray * 3 + 2];
    const float d0 = ds[ray * 3 + 0], d1 = ds[ray * 3 + 1], d2 = ds[ray * 3 + 2];
    const float eps = 1e-9f;
    const float a0 = (fabsf(d0) < eps) ? eps : d0;
    const float a1 = (fabsf(d1) < eps) ? eps : d1;
    const float a2 = (fabsf(d2) < eps) ? eps : d2;

    const float tl0 = (XYZMIN_C - o0) / a0, th0 = (XYZMAX_C - o0) / a0;
    const float tl1 = (XYZMIN_C - o1) / a1, th1 = (XYZMAX_C - o1) / a1;
    const float tl2 = (XYZMIN_C - o2) / a2, th2 = (XYZMAX_C - o2) / a2;
    const float lo0 = fminf(tl0, th0), hi0 = fmaxf(tl0, th0);
    const float lo1 = fminf(tl1, th1), hi1 = fmaxf(tl1, th1);
    const float lo2 = fminf(tl2, th2), hi2 = fmaxf(tl2, th2);
    const float tmin = fmaxf(fmaxf(fmaxf(lo0, lo1), lo2), 0.0f);
    const float tmax = fminf(fminf(hi0, hi1), hi2);
    const bool  hit  = tmax > tmin;

    // plane intersection t's, clipped; pad to 512 with tmax (>= all clipped vals)
    for (int i = tid; i < 512; i += 256) {
        float tv;
        if (i < NPLANES) {
            int axis = i / GRID_N1;
            int p    = i - axis * GRID_N1;
            float oo = (axis == 0) ? o0 : ((axis == 1) ? o1 : o2);
            float aa = (axis == 0) ? a0 : ((axis == 1) ? a1 : a2);
            tv = (XYZMIN_C + VOXEL_SIZE * (float)p - oo) / aa;
            tv = fminf(fmaxf(tv, tmin), tmax);
        } else {
            tv = tmax;
        }
        ts[i] = tv;
    }
    __syncthreads();

    // bitonic sort 512 floats ascending
    for (int k = 2; k <= 512; k <<= 1) {
        for (int j = k >> 1; j > 0; j >>= 1) {
            for (int base = 0; base < 512; base += 256) {
                int i   = base + tid;
                int ixj = i ^ j;
                if (ixj > i) {
                    float x = ts[i], y = ts[ixj];
                    bool up = (i & k) == 0;
                    if ((x > y) == up) { ts[i] = y; ts[ixj] = x; }
                }
            }
            __syncthreads();
        }
    }

    // per-segment features: wave w handles segments w, w+8, ...
    const int wave = tid >> 5;
    const int lane = tid & 31;
    for (int seg = wave; seg < MSEG; seg += 8) {
        const float t_in  = ts[seg];
        const float t_out = ts[seg + 1];
        const bool  mask  = hit && ((t_out - t_in) > 1e-8f);
        const int   row   = ray * MSEG + seg;
        if (lane == 0) {
            maskOut[row] = mask ? 1.0f : 0.0f;
            tinOut[row]  = t_in;
        }
        // entry/exit points in voxel coords (original ds, per reference)
        float pi0 = fminf(fmaxf((o0 + t_in  * d0 - XYZMIN_C) * INV_VS, 0.0f), 128.0f);
        float pi1 = fminf(fmaxf((o1 + t_in  * d1 - XYZMIN_C) * INV_VS, 0.0f), 128.0f);
        float pi2 = fminf(fmaxf((o2 + t_in  * d2 - XYZMIN_C) * INV_VS, 0.0f), 128.0f);
        float po0 = fminf(fmaxf((o0 + t_out * d0 - XYZMIN_C) * INV_VS, 0.0f), 128.0f);
        float po1 = fminf(fmaxf((o1 + t_out * d1 - XYZMIN_C) * INV_VS, 0.0f), 128.0f);
        float po2 = fminf(fmaxf((o2 + t_out * d2 - XYZMIN_C) * INV_VS, 0.0f), 128.0f);
        float md0 = 0.5f * (pi0 + po0), md1 = 0.5f * (pi1 + po1), md2 = 0.5f * (pi2 + po2);
        int ci = (int)fminf(fmaxf(floorf(md0), 0.0f), 127.0f);
        int cj = (int)fminf(fmaxf(floorf(md1), 0.0f), 127.0f);
        int ck = (int)fminf(fmaxf(floorf(md2), 0.0f), 127.0f);
        float A0 = pi0 - (float)ci, A1 = pi1 - (float)cj, A2 = pi2 - (float)ck;
        float B0 = po0 - (float)ci, B1 = po1 - (float)cj, B2 = po2 - (float)ck;
        float M0 = 0.5f * (A0 + B0), M1 = 0.5f * (A1 + B1), M2 = 0.5f * (A2 + B2);
        float e0 = po0 - pi0, e1 = po1 - pi1, e2 = po2 - pi2;
        float seg_len = sqrtf(e0 * e0 + e1 * e1 + e2 * e2) * VOXEL_SIZE;

        float f0 = 0.0f, f1 = 0.0f;  // lane handles dims lane and lane+32
#pragma unroll
        for (int cc = 0; cc < 8; ++cc) {
            const int dz = cc & 1, dy = (cc >> 1) & 1, dx = (cc >> 2) & 1;
            float wA = (dx ? A0 : 1.0f - A0) * (dy ? A1 : 1.0f - A1) * (dz ? A2 : 1.0f - A2);
            float wM = (dx ? M0 : 1.0f - M0) * (dy ? M1 : 1.0f - M1) * (dz ? M2 : 1.0f - M2);
            float wB = (dx ? B0 : 1.0f - B0) * (dy ? B1 : 1.0f - B1) * (dz ? B2 : 1.0f - B2);
            float w  = (wA + 4.0f * wM + wB) * (1.0f / 6.0f);
            size_t idx = ((size_t)((ci + dx) * GRID_N1 + (cj + dy)) * GRID_N1 + (ck + dz))
                         * (size_t)VOXEL_DIM;
            f0 += w * voxels[idx + lane];
            f1 += w * voxels[idx + lane + 32];
        }
        f0 *= seg_len;
        f1 *= seg_len;
        if (!mask) { f0 = 0.0f; f1 = 0.0f; }
        feat16[(size_t)row * 64 + lane]      = (half_t)f0;
        feat16[(size_t)row * 64 + lane + 32] = (half_t)f1;
    }
}

// =====================================================================
// Kernel 2: fused MLP1 + posenc + MLP2 via WMMA.
// 256 thr = 8 waves; each wave owns a 16-row tile (block = 128 rows).
// Weights are pre-shuffled into B-fragment order in LDS (vector loads).
// =====================================================================
__global__ void __launch_bounds__(256) mlp_kernel(
    const half_t* __restrict__ feat16, const float* __restrict__ ds,
    const float* __restrict__ w1, const float* __restrict__ b1,
    const float* __restrict__ w2, const float* __restrict__ b2,
    const float* __restrict__ w3, const float* __restrict__ b3,
    const float* __restrict__ wa, const float* __restrict__ ba,
    const float* __restrict__ wb, const float* __restrict__ bb,
    const float* __restrict__ wc, const float* __restrict__ bc,
    const float* __restrict__ maskBuf,
    float* __restrict__ colorOut, float* __restrict__ sigmaOut) {

    // Fragment-ordered weights (2 ktiles x ntiles x 512 halves each).
    __shared__ __align__(32) half_t sW1[2 * 4 * 512];  // 64x64
    __shared__ __align__(32) half_t sW2[2 * 4 * 512];  // 64x64
    __shared__ __align__(32) half_t sW3[2 * 3 * 512];  // 64x33 -> 48 cols
    __shared__ __align__(32) half_t sWa[2 * 4 * 512];  // 59x64 -> 64 rows
    __shared__ __align__(32) half_t sWb[2 * 4 * 512];  // 64x64
    __shared__ __align__(32) half_t sWc[2 * 1 * 512];  // 64x3  -> 16 cols
    __shared__ float  sB1[64], sB2[64], sB3[48], sBa[64], sBb[64], sBc[16];
    __shared__ __align__(32) half_t sAct[8][16 * 64];

    const int tid = threadIdx.x;
    // ---- stage + convert weights (f32 -> f16) into fragment-ordered LDS ----
    stage_frags(sW1, w1, 64, 64, 64, 4, tid);
    stage_frags(sW2, w2, 64, 64, 64, 4, tid);
    stage_frags(sW3, w3, 33, 64, 33, 3, tid);
    stage_frags(sWa, wa, 64, 59, 64, 4, tid);
    stage_frags(sWb, wb, 64, 64, 64, 4, tid);
    stage_frags(sWc, wc, 3,  64, 3,  1, tid);
    if (tid < 64) { sB1[tid] = b1[tid]; sB2[tid] = b2[tid]; sBa[tid] = ba[tid]; sBb[tid] = bb[tid]; }
    if (tid < 48) sB3[tid] = (tid < 33) ? b3[tid] : 0.0f;
    if (tid < 16) sBc[tid] = (tid < 3) ? bc[tid] : 0.0f;
    __syncthreads();

    const int lane    = tid & 31;
    const int wave    = tid >> 5;
    const int rowBase = blockIdx.x * 128 + wave * 16;
    half_t* act = sAct[wave];

    // ---- stage feature tile (16x64 f16, contiguous) into LDS ----
    {
        const uint32_t* src = (const uint32_t*)(feat16 + (size_t)rowBase * 64);
        uint32_t* dst = (uint32_t*)act;
        for (int it = lane; it < 512; it += 32) dst[it] = src[it];
    }
    __syncthreads();

    // ---- MLP1 layers 1,2 (64->64, ReLU) ----
    dense64(act, sW1, sB1, true, lane);
    __syncthreads();
    dense64(act, sW2, sB2, true, lane);
    __syncthreads();

    // ---- MLP1 layer 3 (64->33, no activation): sigma = col 0, h = cols 1..32 ----
    {
        v16h a0 = load_a(act, 0, lane);
        v16h a1 = load_a(act, 32, lane);
        const int n16 = lane & 15;
        const int kh  = lane >> 4;
#pragma unroll
        for (int j = 0; j < 3; ++j) {
            v8f c = {};
            c = wmma_f16(a0, load_bf(sW3, j,     lane), c);
            c = wmma_f16(a1, load_bf(sW3, 3 + j, lane), c);
            const int n  = 16 * j + n16;
            const float bv = sB3[n];
#pragma unroll
            for (int r = 0; r < 8; ++r) {
                float v = c[r] + bv;
                int m = r + 8 * kh;
                if (n == 0) {
                    int row = rowBase + m;
                    sigmaOut[row] = (maskBuf[row] != 0.0f) ? v : 0.0f;
                } else if (n <= 32) {
                    act[m * 64 + (n - 1)] = (half_t)v;   // h -> input cols 0..31
                }
            }
        }
    }

    // ---- fill MLP2 input cols 32..63: [d(3), sin/cos posenc(24), zeros(5)] ----
    for (int it = lane; it < 512; it += 32) {
        int rrow = it >> 5;
        int col  = 32 + (it & 31);
        float v = 0.0f;
        if (col < 59) {
            int grow = rowBase + rrow;
            int ray  = grow / MSEG;
            float dv0 = ds[ray * 3 + 0], dv1 = ds[ray * 3 + 1], dv2 = ds[ray * 3 + 2];
            int c = col - 32;
            if (c < 3) {
                v = (c == 0) ? dv0 : ((c == 1) ? dv1 : dv2);
            } else {
                int j = c - 3;
                int f = j / 6;
                int r = j - f * 6;
                int comp = (r < 3) ? r : (r - 3);
                float dd  = (comp == 0) ? dv0 : ((comp == 1) ? dv1 : dv2);
                float ang = (float)(1 << f) * dd;
                v = (r < 3) ? __sinf(ang) : __cosf(ang);
            }
        }
        act[rrow * 64 + col] = (half_t)v;
    }
    __syncthreads();

    // ---- MLP2 layers 1,2 (64->64, ReLU) ----
    dense64(act, sWa, sBa, true, lane);
    __syncthreads();
    dense64(act, sWb, sBb, true, lane);
    __syncthreads();

    // ---- MLP2 layer 3 (64->3) + sigmoid + masked color write ----
    {
        v16h a0 = load_a(act, 0, lane);
        v16h a1 = load_a(act, 32, lane);
        v8f c = {};
        c = wmma_f16(a0, load_bf(sWc, 0, lane), c);
        c = wmma_f16(a1, load_bf(sWc, 1, lane), c);
        const int n  = lane & 15;
        const int kh = lane >> 4;
        if (n < 3) {
            float bv = sBc[n];
#pragma unroll
            for (int r = 0; r < 8; ++r) {
                int row = rowBase + r + 8 * kh;
                float v = 1.0f / (1.0f + __expf(-(c[r] + bv)));
                colorOut[row * 3 + n] = (maskBuf[row] != 0.0f) ? v : 0.0f;
            }
        }
    }
}

// =====================================================================
extern "C" void kernel_launch(void* const* d_in, const int* in_sizes, int n_in,
                              void* d_out, int out_size, void* d_ws, size_t ws_size,
                              hipStream_t stream) {
    (void)in_sizes; (void)n_in; (void)out_size; (void)ws_size;
    const float* os     = (const float*)d_in[0];
    const float* ds     = (const float*)d_in[1];
    const float* voxels = (const float*)d_in[2];
    const float* w1 = (const float*)d_in[3];
    const float* b1 = (const float*)d_in[4];
    const float* w2 = (const float*)d_in[5];
    const float* b2 = (const float*)d_in[6];
    const float* w3 = (const float*)d_in[7];
    const float* b3 = (const float*)d_in[8];
    const float* wa = (const float*)d_in[9];
    const float* ba = (const float*)d_in[10];
    const float* wb = (const float*)d_in[11];
    const float* bb = (const float*)d_in[12];
    const float* wc = (const float*)d_in[13];
    const float* bc = (const float*)d_in[14];

    float* out      = (float*)d_out;
    float* colorOut = out;              // [ROWS*3]
    float* sigmaOut = out + OFF_SIGMA;  // [ROWS]
    float* maskOut  = out + OFF_MASK;   // [ROWS]
    float* tinOut   = out + OFF_TIN;    // [ROWS]

    half_t* feat16 = (half_t*)d_ws;     // ROWS*64 f16 features (~101 MB)

    geom_kernel<<<NRAYS, 256, 0, stream>>>(os, ds, voxels, feat16, maskOut, tinOut);
    mlp_kernel<<<ROWS / 128, 256, 0, stream>>>(feat16, ds,
                                               w1, b1, w2, b2, w3, b3,
                                               wa, ba, wb, bb, wc, bc,
                                               maskOut, colorOut, sigmaOut);
}
